// GCN_84353157693520
// MI455X (gfx1250) — matrix operand order repaired and checked by reading, not compiled, not run on hardware
//
#include <hip/hip_runtime.h>

// Problem constants (match reference)
#define NND 100000
#define NED 600000
#define FD  128
#define NG  64

typedef __attribute__((ext_vector_type(16))) __bf16 v16bf;
typedef __attribute__((ext_vector_type(8)))  float  v8f;

// ---------------------------------------------------------------------------
// degree / norm
// ---------------------------------------------------------------------------
__global__ void k_init_deg(float* deg, int n) {
    int i = blockIdx.x * blockDim.x + threadIdx.x;
    if (i < n) deg[i] = 1.0f;            // self-loop contributes 1
}

__global__ void k_count(const int* __restrict__ col, float* deg, int E) {
    int e = blockIdx.x * blockDim.x + threadIdx.x;
    if (e < E) unsafeAtomicAdd(&deg[col[e]], 1.0f);
}

__global__ void k_dinv(float* deg, int n) {
    int i = blockIdx.x * blockDim.x + threadIdx.x;
    if (i < n) deg[i] = rsqrtf(deg[i]);  // in place: deg -> deg^{-1/2}
}

// ---------------------------------------------------------------------------
// Pre-pack fp32 weights into the bf16 WMMA B-operand layout.
// Record = one lane's v16bf for (weight w, col-tile t, k-step s, lane).
// Lane layout (16x16x32 bf16 B, 32x16 KxN):
//   lanes 0-15  : column n = 16t + lane,    K = 32s + 0..7, 16..23  (packed pairs)
//   lanes 16-31 : column n = 16t + lane-16, K = 32s + 8..15, 24..31
// ---------------------------------------------------------------------------
__global__ void k_prep_w(const float* __restrict__ W1,
                         const float* __restrict__ W2,
                         const float* __restrict__ W3,
                         __bf16* __restrict__ Bp) {
    int idx = blockIdx.x * blockDim.x + threadIdx.x;
    if (idx >= 3 * 8 * 4 * 32) return;
    int lane = idx & 31;
    int s    = (idx >> 5) & 3;
    int t    = (idx >> 7) & 7;
    int w    = idx >> 10;
    const float* W = (w == 0) ? W1 : (w == 1) ? W2 : W3;
    int n  = t * 16 + (lane & 15);
    int kb = 32 * s + ((lane >> 4) ? 8 : 0);
    __bf16* dst = Bp + (size_t)w * (FD * FD) + ((size_t)((t * 4 + s) * 32 + lane)) * 16;
#pragma unroll
    for (int j = 0; j < 16; ++j)
        dst[j] = (__bf16)W[(size_t)(kb + (j & 7) + ((j >> 3) << 4)) * FD + n];
}

// ---------------------------------------------------------------------------
// GEMM: T[N,128] = X[N,128] @ W[128,128]  via v_wmma_f32_16x16x32_bf16
// One wave per 16-row block; 8 col tiles x 4 k-steps = 32 WMMAs per wave.
// ---------------------------------------------------------------------------
__global__ __launch_bounds__(256) void k_gemm_wmma(
    const float* __restrict__ X, const __bf16* __restrict__ Bp,
    float* __restrict__ T, int nRB) {
    int wave = threadIdx.x >> 5;
    int lane = threadIdx.x & 31;
    int rb = blockIdx.x * 8 + wave;
    if (rb >= nRB) return;                     // wave-uniform: EXEC stays all-1s

    int m  = lane & 15;                        // row within 16-row tile
    int hi = lane >> 4;                        // lane group (0: lanes 0-15)
    const float* xrow = X + (size_t)(rb * 16 + m) * FD;

    // A operand, per ISA 16-bit A 16x32 layout:
    //   elems 0..7  = K kb..kb+7,  elems 8..15 = K kb+16..kb+23,  kb = 32s + 8*hi
    v16bf a[4];
#pragma unroll
    for (int s = 0; s < 4; ++s) {
        int kb = 32 * s + hi * 8;
        float4 f0 = *(const float4*)(xrow + kb);
        float4 f1 = *(const float4*)(xrow + kb + 4);
        float4 f2 = *(const float4*)(xrow + kb + 16);
        float4 f3 = *(const float4*)(xrow + kb + 20);
        a[s][0]  = (__bf16)f0.x; a[s][1]  = (__bf16)f0.y;
        a[s][2]  = (__bf16)f0.z; a[s][3]  = (__bf16)f0.w;
        a[s][4]  = (__bf16)f1.x; a[s][5]  = (__bf16)f1.y;
        a[s][6]  = (__bf16)f1.z; a[s][7]  = (__bf16)f1.w;
        a[s][8]  = (__bf16)f2.x; a[s][9]  = (__bf16)f2.y;
        a[s][10] = (__bf16)f2.z; a[s][11] = (__bf16)f2.w;
        a[s][12] = (__bf16)f3.x; a[s][13] = (__bf16)f3.y;
        a[s][14] = (__bf16)f3.z; a[s][15] = (__bf16)f3.w;
    }

    const v16bf* bp = (const v16bf*)Bp;        // records of 16 bf16, 32B aligned
#pragma unroll
    for (int t = 0; t < 8; ++t) {
        v8f acc = {};
#pragma unroll
        for (int s = 0; s < 4; ++s) {
            v16bf b = bp[(size_t)(t * 4 + s) * 32 + lane];
            acc = __builtin_amdgcn_wmma_f32_16x16x32_bf16(
                false, a[s], false, b, (short)0, acc, false, false);
        }
        // D layout: VGPR v -> M = v + 8*hi, N = m (within tile t)
#pragma unroll
        for (int v = 0; v < 8; ++v)
            T[(size_t)(rb * 16 + hi * 8 + v) * FD + t * 16 + m] = acc[v];
    }
}

// ---------------------------------------------------------------------------
// message passing / elementwise
// ---------------------------------------------------------------------------
__global__ void k_zero(float* p, size_t n) {
    size_t i = (size_t)blockIdx.x * blockDim.x + threadIdx.x;
    if (i < n) p[i] = 0.0f;
}

__global__ void k_scatter(const int* __restrict__ row, const int* __restrict__ col,
                          const float* __restrict__ dinv, const float* __restrict__ T,
                          float* __restrict__ H, int E) {
    size_t idx = (size_t)blockIdx.x * blockDim.x + threadIdx.x;
    if (idx >= (size_t)E * FD) return;
    int e = (int)(idx >> 7);
    int f = (int)(idx & (FD - 1));
    int r = row[e], c = col[e];
    float nrm = dinv[r] * dinv[c];
    unsafeAtomicAdd(&H[(size_t)c * FD + f], nrm * T[(size_t)r * FD + f]);
}

__global__ void k_finish(const float* __restrict__ T, const float* __restrict__ dinv,
                         const float* __restrict__ b, float* __restrict__ H,
                         int n, int relu) {
    size_t idx = (size_t)blockIdx.x * blockDim.x + threadIdx.x;
    if (idx >= (size_t)n * FD) return;
    int node = (int)(idx >> 7);
    int f    = (int)(idx & (FD - 1));
    float d  = dinv[node];
    float v  = H[idx] + d * d * T[idx] + b[f];
    H[idx]   = relu ? fmaxf(v, 0.0f) : v;
}

// ---------------------------------------------------------------------------
// pooling + head
// ---------------------------------------------------------------------------
__global__ void k_pool(const float* __restrict__ H, const int* __restrict__ batch,
                       float* __restrict__ pooled, float* __restrict__ counts, int n) {
    size_t idx = (size_t)blockIdx.x * blockDim.x + threadIdx.x;
    if (idx >= (size_t)n * FD) return;
    int node = (int)(idx >> 7);
    int f    = (int)(idx & (FD - 1));
    int g    = batch[node];
    unsafeAtomicAdd(&pooled[(size_t)g * FD + f], H[idx]);
    if (f == 0) unsafeAtomicAdd(&counts[g], 1.0f);
}

__global__ void k_head(const float* __restrict__ pooled, const float* __restrict__ counts,
                       const float* __restrict__ Wl, const float* __restrict__ bl,
                       float* __restrict__ out) {
    int t = threadIdx.x;            // 128 threads = 64 graphs x 2 classes
    int g = t >> 1, c = t & 1;
    float inv = 1.0f / fmaxf(counts[g], 1.0f);
    float acc = bl[c];
#pragma unroll 4
    for (int f = 0; f < FD; ++f)
        acc += (pooled[(size_t)g * FD + f] * inv) * Wl[f * 2 + c];
    out[g * 2 + c] = acc;
}

// ---------------------------------------------------------------------------
// launch
// ---------------------------------------------------------------------------
extern "C" void kernel_launch(void* const* d_in, const int* in_sizes, int n_in,
                              void* d_out, int out_size, void* d_ws, size_t ws_size,
                              hipStream_t stream) {
    const float* x     = (const float*)d_in[0];
    const int*   ei    = (const int*)d_in[1];     // [2, E]
    const int*   batch = (const int*)d_in[2];
    const float* W1 = (const float*)d_in[3];
    const float* b1 = (const float*)d_in[4];
    const float* W2 = (const float*)d_in[5];
    const float* b2 = (const float*)d_in[6];
    const float* W3 = (const float*)d_in[7];
    const float* b3 = (const float*)d_in[8];
    const float* Wl = (const float*)d_in[9];
    const float* bl = (const float*)d_in[10];

    const int E = in_sizes[1] / 2;   // 600000
    const int N = in_sizes[2];       // 100000 (divisible by 16)

    // workspace layout (floats; ~103 MB total)
    float*  T      = (float*)d_ws;                  // [N,128] GEMM output
    float*  H      = T + (size_t)N * FD;            // [N,128] aggregated output
    float*  dinv   = H + (size_t)N * FD;            // [N] deg -> deg^{-1/2}
    float*  pooled = dinv + N;                      // [64,128]
    float*  counts = pooled + (size_t)NG * FD;      // [64]
    __bf16* Bp     = (__bf16*)(counts + NG);        // 3 x 128*128 packed bf16 (32B aligned)

    const int* row = ei;
    const int* col = ei + E;

    // norms + weight packing
    k_init_deg<<<(N + 255) / 256, 256, 0, stream>>>(dinv, N);
    k_count<<<(E + 255) / 256, 256, 0, stream>>>(col, dinv, E);
    k_dinv<<<(N + 255) / 256, 256, 0, stream>>>(dinv, N);
    k_prep_w<<<(3 * 8 * 4 * 32 + 255) / 256, 256, 0, stream>>>(W1, W2, W3, Bp);

    const int nRB = N / 16;                          // 6250 row blocks
    dim3 gemmGrid((nRB + 7) / 8);                    // 8 waves / block
    const size_t NF = (size_t)N * FD;
    const int nfBlocks = (int)((NF + 255) / 256);
    const size_t EF = (size_t)E * FD;
    const int efBlocks = (int)((EF + 255) / 256);

    const float* layerIn = x;
    const float* biases[3] = {b1, b2, b3};
    for (int L = 0; L < 3; ++L) {
        k_gemm_wmma<<<gemmGrid, 256, 0, stream>>>(layerIn, Bp + (size_t)L * FD * FD, T, nRB);
        k_zero<<<nfBlocks, 256, 0, stream>>>(H, NF);
        k_scatter<<<efBlocks, 256, 0, stream>>>(row, col, dinv, T, H, E);
        k_finish<<<nfBlocks, 256, 0, stream>>>(T, dinv, biases[L], H, N, (L < 2) ? 1 : 0);
        layerIn = H;
    }

    k_zero<<<(NG * FD + NG + 255) / 256, 256, 0, stream>>>(pooled, (size_t)NG * FD + NG);
    k_pool<<<nfBlocks, 256, 0, stream>>>(H, batch, pooled, counts, N);
    k_head<<<1, NG * 2, 0, stream>>>(pooled, counts, Wl, bl, (float*)d_out);
}
